// SSD_cell_27625229647900
// MI455X (gfx1250) — matrix operand order repaired and checked by reading, not compiled, not run on hardware
//
#include <hip/hip_runtime.h>
#include <math.h>

typedef __attribute__((ext_vector_type(16))) _Float16 v16h;
typedef __attribute__((ext_vector_type(8)))  _Float16 v8h;
typedef __attribute__((ext_vector_type(8)))  float    v8f;

#define T_LEN   4096
#define BATCH   8
#define SPATIAL 16
#define BS      128       // BATCH*SPATIAL sequences
#define CIN     128
#define COUT    193
#define COP     208       // padded to 13 tiles of 16
#define KW      4
#define RDIM    512       // CIN*KW reduction
#define NDIM    64
#define CHUNK   64
#define NCHUNK  64
#define TPAD    4099      // T + (KW-1) leading zero rows
#define TB      128       // timesteps per K2 block
#define XROWS   131       // TB + KW - 1
#define XSTR    136       // padded halves per LDS row (272B: 16B-aligned, bank-rotating)

// ---------------- TDM support (Tensor Data Mover) ----------------
#if __has_builtin(__builtin_amdgcn_tensor_load_to_lds)
#define HAVE_TDM 1
typedef __attribute__((ext_vector_type(4))) unsigned int v4u;
typedef __attribute__((ext_vector_type(8))) int          v8i_;
typedef __attribute__((ext_vector_type(4))) int          v4i_;

// 1-D contiguous global -> LDS with periodic padding inserted on the LDS side.
// n8 = number of 8-byte elements; pad_int_code: ISA pad_interval (0=2,..,4=32,5=64 dwords);
// pad_amt_code: ISA pad_amount (3 = 4 dwords = 16B).
__device__ inline void tdm_load_1d(unsigned lds_off, const void* gaddr, unsigned n8,
                                   unsigned pad_int_code, unsigned pad_amt_code) {
  unsigned long long ga = (unsigned long long)(uintptr_t)gaddr;
  v4u g0;
  g0[0] = 1u;                                              // count=1, user mode
  g0[1] = lds_off;                                         // lds_addr
  g0[2] = (unsigned)ga;                                    // global_addr[31:0]
  g0[3] = (unsigned)((ga >> 32) & 0x01FFFFFFu) | (2u << 30); // addr[56:32] | type=2
  v8i_ g1;
  g1[0] = (int)((3u << 16) | (1u << 20) |                  // data_size=8B, pad_enable
                (pad_int_code << 22) | (pad_amt_code << 25));
  g1[1] = (int)((n8 & 0xFFFFu) << 16);                     // tensor_dim0[15:0]
  g1[2] = (int)((n8 >> 16) | (1u << 16));                  // tensor_dim0[31:16], tensor_dim1=1
  g1[3] = (int)(n8 << 16);                                 // tile_dim0
  g1[4] = 0;                                               // tile_dim1/2 unused
  g1[5] = (int)n8;                                         // tensor_dim0_stride
  g1[6] = 0; g1[7] = 0;
  v4i_ z4 = {0, 0, 0, 0};
  v8i_ z8 = {0, 0, 0, 0, 0, 0, 0, 0};
  __builtin_amdgcn_tensor_load_to_lds(g0, g1, z4, z4, z8, 0);   // 6-arg toolchain form
}
#define TDM_WAIT() __builtin_amdgcn_s_wait_tensorcnt((short)0)
#else
#define HAVE_TDM 0
#endif

// ---------- helpers ----------
__device__ inline v16h pack16(const _Float16* p0, const _Float16* p1) {
  v8h a = *(const v8h*)p0;
  v8h b = *(const v8h*)p1;
  v16h r;
#pragma unroll
  for (int i = 0; i < 8; ++i) { r[i] = a[i]; r[i + 8] = b[i]; }
  return r;
}

// ---------- K0: conv weights -> f16, r = k*128+ci order, zero-pad co>=193 ----------
__global__ void k0_prep_w(const float* __restrict__ conv_w, _Float16* __restrict__ Wh) {
  int o = blockIdx.x * blockDim.x + threadIdx.x;   // 0 .. 208*512-1
  int co = o >> 9;
  int r  = o & 511;
  int k  = r >> 7;
  int ci = r & 127;
  float v = (co < COUT) ? conv_w[(co * CIN + ci) * KW + k] : 0.f;
  Wh[o] = (_Float16)v;
}

// ---------- K1: x -> f16 padded [bs][t+3][ci]; fused sigmoid(Dx) ----------
__global__ void __launch_bounds__(128)
k1_prep_x(const float* __restrict__ x, const float* __restrict__ pass_w,
          const float* __restrict__ pass_b,
          _Float16* __restrict__ xhp, float* __restrict__ sigDx) {
  int t = blockIdx.x, bs = blockIdx.y, ci = threadIdx.x;
  int b = bs >> 4, s = bs & 15;
  float xv = x[(((size_t)t * BATCH + b) * SPATIAL + s) * CIN + ci];
  xhp[((size_t)bs * TPAD + (t + 3)) * CIN + ci] = (_Float16)xv;
  if (t < 3) xhp[((size_t)bs * TPAD + t) * CIN + ci] = (_Float16)0.f;

  float p = xv * pass_w[ci];
#pragma unroll
  for (int o = 16; o >= 1; o >>= 1) p += __shfl_xor(p, o, 32);
  __shared__ float red[4];
  int wid = threadIdx.x >> 5, lane = threadIdx.x & 31;
  if (lane == 0) red[wid] = p;
  __syncthreads();
  if (threadIdx.x == 0) {
    float d = red[0] + red[1] + red[2] + red[3] + pass_b[0];
    sigDx[(size_t)bs * T_LEN + t] = 1.f / (1.f + __expf(-d));
  }
}

// ---------- K2: implicit-GEMM conv via WMMA, x slab staged in LDS by TDM ----------
__global__ void __launch_bounds__(256)
k2_conv_wmma(const _Float16* __restrict__ Wh, const _Float16* __restrict__ xhp,
             const float* __restrict__ conv_b, const float* __restrict__ reduce_w,
             const float* __restrict__ dt_param,
             _Float16* __restrict__ Bmh, _Float16* __restrict__ Cmh,
             float* __restrict__ lAdb, float* __restrict__ sBb,
             float* __restrict__ xwp /* [5][bs][t] partials */) {
  __shared__ __attribute__((aligned(16))) _Float16 Xs[XROWS][XSTR];  // 35632 B

  int tb = blockIdx.x;          // t-block (0..31)
  int bs = blockIdx.y;          // sequence (0..127)
  int tid = threadIdx.x;
  int wid = tid >> 5, lane = tid & 31;
  int tloc = lane & 15, hi = lane >> 4;

  const _Float16* gx = xhp + ((size_t)bs * TPAD + (size_t)tb * TB) * CIN;

#if HAVE_TDM
  if (wid == 0) {
    // 131 rows x 256B contiguous = 4192 x 8B; pad 4 dwords after every 64 dwords
    tdm_load_1d((unsigned)(uintptr_t)&Xs[0][0], gx, XROWS * (CIN / 4), 5u, 3u);
    TDM_WAIT();
  }
  __syncthreads();
#else
  for (int e = tid; e < XROWS * 16; e += 256) {
    int row = e >> 4, q = e & 15;
    *(v8h*)&Xs[row][q * 8] = *(const v8h*)(gx + (size_t)row * CIN + q * 8);
  }
  __syncthreads();
#endif

  // 104 output tiles (13 co-tiles x 8 t-tiles); each of 8 waves takes 13
  for (int tile = wid; tile < 104; tile += 8) {
    int ct = tile % 13;
    int tt = tile / 13;
    int t  = tb * TB + tt * 16 + tloc;
    int trow = tt * 16 + tloc;

    // A rows: M = lane&15, K chunks {hi*8+0..7, hi*8+16..23} per 32-tile
    const _Float16* Wrow = Wh + (size_t)(ct * 16 + tloc) * RDIM + hi * 8;

    v8f acc = {};
#pragma unroll
    for (int ks = 0; ks < 16; ++ks) {
      int r0  = ks * 32;
      int kk  = r0 >> 7;          // conv tap (r = k*128+ci)
      int ci0 = r0 & 127;
      v16h aF = pack16(Wrow + r0, Wrow + r0 + 16);
      const _Float16* xp = &Xs[trow + kk][ci0 + hi * 8];
      v16h bF = pack16(xp, xp + 16);
      acc = __builtin_amdgcn_wmma_f32_16x16x32_f16(false, aF, false, bF,
                                                   (short)0, acc, false, false);
    }

    // epilogue: lane holds column t, rows co = ct*16 + hi*8 + v
    size_t base = ((size_t)bs * T_LEN + t) * 64;
    if (ct < 8) {                          // Bm (tiles 0-3) / Cm (tiles 4-7)
      v8h hv;
#pragma unroll
      for (int v = 0; v < 8; ++v) {
        int co = ct * 16 + hi * 8 + v;
        float z = acc[v] + conv_b[co];
        hv[v] = (_Float16)(z + 1.f / (1.f + __expf(-z)));
      }
      int n0 = ct * 16 + hi * 8;
      if (ct < 4) *(v8h*)(Bmh + base + n0) = hv;
      else        *(v8h*)(Cmh + base + (n0 - 64)) = hv;
    } else {                               // tiles 8-12: A channel + xc channels
      float dt = logf(1.f + __expf(dt_param[0])) + 0.01f;
      float wsum = 0.f;
#pragma unroll
      for (int v = 0; v < 8; ++v) {
        int co = ct * 16 + hi * 8 + v;
        if (co > 192) continue;            // padded rows
        float z = acc[v] + conv_b[co];
        float a = z + 1.f / (1.f + __expf(-z));
        if (co == 128) {                   // A channel
          float l = dt * a;
          lAdb[(size_t)bs * T_LEN + t] = l;
          sBb [(size_t)bs * T_LEN + t] = (__expf(l) - 1.f) / a;
        } else {                           // xc: p = co-129, fold reduce_w
          wsum += a * reduce_w[co - 129];
        }
      }
      wsum += __shfl_xor(wsum, 16, 32);    // combine hi halves for this t
      if (hi == 0)
        xwp[((size_t)(ct - 8) * BS + bs) * T_LEN + t] = wsum;
    }
  }
}

// ---------- K4: per (bs, chunk): G = C.B^T (WMMA), decay mask, ydiag, cs ----------
__global__ void __launch_bounds__(32)
k4_chunk(const _Float16* __restrict__ Bmh, const _Float16* __restrict__ Cmh,
         const float* __restrict__ lAdb, const float* __restrict__ sBb,
         const float* __restrict__ xwp,
         float* __restrict__ ydiag, float* __restrict__ csb,
         float* __restrict__ alastb) {
  int c = blockIdx.x, bs = blockIdx.y;
  int lane = threadIdx.x;
  int tloc = lane & 15, hi = lane >> 4;
  size_t tbase = (size_t)bs * T_LEN + (size_t)c * CHUNK;

  __shared__ __attribute__((aligned(16))) _Float16 Bl[64][72];  // 144B stride
  __shared__ __attribute__((aligned(16))) _Float16 Cl[64][72];
  __shared__ float Ac[64];
  __shared__ float uu[64];
  __shared__ float w2[64];

#if HAVE_TDM
  // 64 rows x 128B contiguous = 1024 x 8B; pad 4 dwords after every 32 dwords
  tdm_load_1d((unsigned)(uintptr_t)&Bl[0][0], Bmh + tbase * 64, 1024u, 4u, 3u);
  tdm_load_1d((unsigned)(uintptr_t)&Cl[0][0], Cmh + tbase * 64, 1024u, 4u, 3u);
  TDM_WAIT();   // single wave: data visible once TENSORcnt drains
#else
#pragma unroll
  for (int rr = 0; rr < 2; ++rr) {
    int row = lane + rr * 32;
    const _Float16* bp = Bmh + (tbase + row) * 64;
    const _Float16* cp = Cmh + (tbase + row) * 64;
#pragma unroll
    for (int q = 0; q < 8; ++q) {
      *(v8h*)&Bl[row][q * 8] = *(const v8h*)(bp + q * 8);
      *(v8h*)&Cl[row][q * 8] = *(const v8h*)(cp + q * 8);
    }
  }
#endif

  // inclusive wave scan of lAd over the chunk
  float a0 = lAdb[tbase + lane];
  float a1 = lAdb[tbase + 32 + lane];
  float s0 = a0;
#pragma unroll
  for (int o = 1; o < 32; o <<= 1) { float nv = __shfl_up(s0, o, 32); if (lane >= o) s0 += nv; }
  float tot0 = __shfl(s0, 31, 32);
  float s1 = a1;
#pragma unroll
  for (int o = 1; o < 32; o <<= 1) { float nv = __shfl_up(s1, o, 32); if (lane >= o) s1 += nv; }
  s1 += tot0;
  float alast = __shfl(s1, 31, 32);
  Ac[lane] = s0; Ac[lane + 32] = s1;

  size_t xwstride = (size_t)BS * T_LEN;
#pragma unroll
  for (int rr = 0; rr < 2; ++rr) {
    int tl = lane + rr * 32;
    float xv = 0.f;
#pragma unroll
    for (int j = 0; j < 5; ++j) xv += xwp[(size_t)j * xwstride + tbase + tl];
    float u  = xv * sBb[tbase + tl];     // fold (e^lAd - 1)/A scale into xw
    float ac = rr ? s1 : s0;
    uu[tl] = u;
    w2[tl] = __expf(alast - ac) * u;     // decay-to-chunk-end * u
  }
  if (lane == 0) alastb[(size_t)bs * NCHUNK + c] = alast;
  __syncthreads();

  // G = C * B^T over n (K=64 -> 2 wmma per tile), mask with L, reduce with u
  for (int lt = 0; lt < 4; ++lt) {
    float yacc[8];
#pragma unroll
    for (int v = 0; v < 8; ++v) yacc[v] = 0.f;
    int l0 = lt * 16 + tloc;
    float acl[8];
#pragma unroll
    for (int v = 0; v < 8; ++v) acl[v] = Ac[lt * 16 + hi * 8 + v];
    for (int mt = 0; mt < 4; ++mt) {
      int m0 = mt * 16 + tloc;
      v8f acc = {};
#pragma unroll
      for (int ks = 0; ks < 2; ++ks) {
        v16h aF = pack16(&Cl[l0][ks * 32 + hi * 8],  &Cl[l0][ks * 32 + hi * 8 + 16]);
        v16h bF = pack16(&Bl[m0][ks * 32 + hi * 8], &Bl[m0][ks * 32 + hi * 8 + 16]);
        acc = __builtin_amdgcn_wmma_f32_16x16x32_f16(false, aF, false, bF,
                                                     (short)0, acc, false, false);
      }
      int m = mt * 16 + tloc;
      float acm = Ac[m];
      float um  = uu[m];
#pragma unroll
      for (int v = 0; v < 8; ++v) {
        int l = lt * 16 + hi * 8 + v;
        float p = (m <= l) ? acc[v] * __expf(acl[v] - acm) * um : 0.f;
#pragma unroll
        for (int o = 1; o < 16; o <<= 1) p += __shfl_xor(p, o, 32);
        yacc[v] += p;
      }
    }
    if (tloc == 0) {
#pragma unroll
      for (int v = 0; v < 8; ++v)
        ydiag[tbase + lt * 16 + hi * 8 + v] = yacc[v];
    }
  }

  // cs[n] = sum_l B[l,n] * w2[l]  (chunk contribution to scanned state)
#pragma unroll
  for (int rr = 0; rr < 2; ++rr) {
    int n = lane + rr * 32;
    float ssum = 0.f;
#pragma unroll 8
    for (int l2 = 0; l2 < 64; ++l2) ssum += (float)Bl[l2][n] * w2[l2];
    csb[((size_t)bs * NCHUNK + c) * NDIM + n] = ssum;
  }
}

// ---------- K5: sequential chunk scan ----------
__global__ void k5_scan(const float* __restrict__ csb, const float* __restrict__ alastb,
                        float* __restrict__ swin) {
  int id = blockIdx.x * blockDim.x + threadIdx.x;   // 8192 = BS*NDIM
  int bs = id >> 6, n = id & 63;
  float sw = 0.f;
  for (int c = 0; c < NCHUNK; ++c) {
    size_t o = ((size_t)bs * NCHUNK + c) * NDIM + n;
    swin[o] = sw;                                   // entering state
    sw = __expf(alastb[(size_t)bs * NCHUNK + c]) * sw + csb[o];
  }
}

// ---------- K6: yoff + combine + output ----------
__global__ void __launch_bounds__(64)
k6_out(const _Float16* __restrict__ Cmh, const float* __restrict__ lAdb,
       const float* __restrict__ swin, const float* __restrict__ ydiag,
       const float* __restrict__ sigDx, const float* __restrict__ reduce_b,
       float* __restrict__ out) {
  int c = blockIdx.x, bs = blockIdx.y, l = threadIdx.x;
  size_t tbase = (size_t)bs * T_LEN + (size_t)c * CHUNK;
  __shared__ float la[64], sw[64];
  la[l] = lAdb[tbase + l];
  sw[l] = swin[((size_t)bs * NCHUNK + c) * NDIM + l];
  __syncthreads();
  float ac = 0.f;
  for (int i = 0; i <= l; ++i) ac += la[i];
  const _Float16* crow = Cmh + (tbase + l) * 64;
  float dot = 0.f;
#pragma unroll 8
  for (int n = 0; n < 64; ++n) dot += (float)crow[n] * sw[n];
  int t = c * CHUNK + l;
  float y = ydiag[tbase + l] + __expf(ac) * dot + reduce_b[0] + sigDx[tbase + l];
  int b = bs >> 4, s = bs & 15;
  out[((size_t)t * BATCH + b) * SPATIAL + s] = y;
}

// ---------- launch ----------
extern "C" void kernel_launch(void* const* d_in, const int* in_sizes, int n_in,
                              void* d_out, int out_size, void* d_ws, size_t ws_size,
                              hipStream_t stream) {
  (void)in_sizes; (void)n_in; (void)out_size; (void)ws_size;
  const float* x        = (const float*)d_in[0];
  const float* conv_w   = (const float*)d_in[1];
  const float* conv_b   = (const float*)d_in[2];
  const float* pass_w   = (const float*)d_in[3];
  const float* pass_b   = (const float*)d_in[4];
  const float* reduce_w = (const float*)d_in[5];
  const float* reduce_b = (const float*)d_in[6];
  const float* dt_param = (const float*)d_in[7];
  float* out = (float*)d_out;

  char* w = (char*)d_ws;
  auto take = [&](size_t bytes) -> char* {
    char* p = w; w += (bytes + 255) & ~(size_t)255; return p;
  };
  _Float16* Wh    = (_Float16*)take((size_t)COP * RDIM * 2);
  _Float16* xhp   = (_Float16*)take((size_t)BS * TPAD * CIN * 2);
  float*    sigDx = (float*)take((size_t)BS * T_LEN * 4);
  float*    lAdb  = (float*)take((size_t)BS * T_LEN * 4);
  float*    sBb   = (float*)take((size_t)BS * T_LEN * 4);
  float*    xwp   = (float*)take((size_t)5 * BS * T_LEN * 4);
  _Float16* Bmh   = (_Float16*)take((size_t)BS * T_LEN * 64 * 2);
  _Float16* Cmh   = (_Float16*)take((size_t)BS * T_LEN * 64 * 2);
  float*    ydiag = (float*)take((size_t)BS * T_LEN * 4);
  float*    csb   = (float*)take((size_t)BS * NCHUNK * NDIM * 4);
  float*    alast = (float*)take((size_t)BS * NCHUNK * 4);
  float*    swin  = (float*)take((size_t)BS * NCHUNK * NDIM * 4);

  k0_prep_w<<<(COP * RDIM) / 256, 256, 0, stream>>>(conv_w, Wh);
  k1_prep_x<<<dim3(T_LEN, BS), 128, 0, stream>>>(x, pass_w, pass_b, xhp, sigDx);
  k2_conv_wmma<<<dim3(T_LEN / TB, BS), 256, 0, stream>>>(Wh, xhp, conv_b, reduce_w,
                                                         dt_param, Bmh, Cmh, lAdb, sBb, xwp);
  k4_chunk<<<dim3(NCHUNK, BS), 32, 0, stream>>>(Bmh, Cmh, lAdb, sBb, xwp, ydiag, csb, alast);
  k5_scan<<<32, 256, 0, stream>>>(csb, alast, swin);
  k6_out<<<dim3(NCHUNK, BS), 64, 0, stream>>>(Cmh, lAdb, swin, ydiag, sigDx, reduce_b, out);
}
